// MultiHeadAttention_61787399520573
// MI455X (gfx1250) — matrix operand order repaired and checked by reading, not compiled, not run on hardware
//
#include <hip/hip_runtime.h>

// ---------------------------------------------------------------------------
// MI455X (gfx1250) multi-head attention: bf16 WMMA pipeline with TDM staging
//   x,W -> bf16 -> QKV gemm (wmma + tensor_load_to_lds) -> flash attention
//   (wmma) -> proj gemm (wmma + TDM)
// ---------------------------------------------------------------------------

typedef __bf16 bf16x16 __attribute__((ext_vector_type(16)));
typedef float  f32x8   __attribute__((ext_vector_type(8)));
typedef unsigned int u32x4 __attribute__((ext_vector_type(4)));
typedef int          i32x4 __attribute__((ext_vector_type(4)));
typedef int          i32x8 __attribute__((ext_vector_type(8)));

union Frag {
    bf16x16        v;
    unsigned int   u[8];
    unsigned short s[16];
};

__device__ __forceinline__ unsigned short f32_to_bf16(float f) {
    unsigned int u = __float_as_uint(f);
    u += 0x7FFFu + ((u >> 16) & 1u);          // round-to-nearest-even
    return (unsigned short)(u >> 16);
}

// ---------------------------------------------------------------------------
// Tensor Data Mover: issue a 2-D tile load (bf16 elements) into LDS.
// D# per CDNA5 ISA 8.3/8.4: group0 = {count, lds_addr, global_addr, type=2},
// group1 = {data_size=2B, tensor_dim0/1, tile_dim0/1, tensor_dim0_stride}.
// ---------------------------------------------------------------------------
__device__ __forceinline__ void tdm_load_tile_2d(
        unsigned lds_byte_off, const unsigned short* gsrc,
        int tile_rows, int tile_cols, int tensor_rows, int tensor_cols)
{
    unsigned long long ga = (unsigned long long)(size_t)gsrc;
    u32x4 g0;
    g0[0] = 1u;                                       // count=1, user desc
    g0[1] = lds_byte_off;                             // lds_addr
    g0[2] = (unsigned)ga;                             // global_addr[31:0]
    g0[3] = (unsigned)((ga >> 32) & 0x01FFFFFFu)      // global_addr[56:32]
          | (2u << 30);                               // type = 2 ("image")
    i32x8 g1;
    g1[0] = 1 << 16;                                  // data_size=1 -> 2 bytes
    g1[1] = (int)(((unsigned)tensor_cols & 0xFFFFu) << 16);        // dim0 lo16
    g1[2] = (int)(((unsigned)tensor_cols >> 16)
          | (((unsigned)tensor_rows & 0xFFFFu) << 16));            // dim0hi|dim1lo
    g1[3] = (int)(((unsigned)tensor_rows >> 16)
          | ((unsigned)tile_cols << 16));                          // dim1hi|tile0
    g1[4] = tile_rows & 0xFFFF;                       // tile_dim1 (tile_dim2=0)
    g1[5] = tensor_cols;                              // dim0_stride lo (row pitch)
    g1[6] = 0;
    g1[7] = 0;
    i32x4 z4 = {0, 0, 0, 0};
#if defined(__clang_major__) && (__clang_major__ >= 23)
    i32x8 z8 = {0, 0, 0, 0, 0, 0, 0, 0};
    __builtin_amdgcn_tensor_load_to_lds(g0, g1, z4, z4, z8, 0);
#else
    __builtin_amdgcn_tensor_load_to_lds(g0, g1, z4, z4, 0);
#endif
}

// ---------------------------------------------------------------------------
// elementwise f32 -> bf16
// ---------------------------------------------------------------------------
__global__ void cvt_bf16_kernel(const float* __restrict__ in,
                                unsigned short* __restrict__ out, int n) {
    int i = blockIdx.x * blockDim.x + threadIdx.x;
    if (i < n) out[i] = f32_to_bf16(in[i]);
}

// ---------------------------------------------------------------------------
// C[M,Nout] = A_bf16[M,K] @ W_bf16[Nout,K]^T + bias
// mode 0: store f32 to outF;  mode 1: scatter q/k/v [B,H,N,hd] bf16 (q*0.125)
// block = 128 threads (4 waves 2x2). Block tile 128x128, wave tile 64x64
// (4x4 WMMA, 128 acc VGPRs). LDS double-buffered, filled by TDM from wave 0,
// synchronized with s_wait_tensorcnt + workgroup barrier.
// ---------------------------------------------------------------------------
__global__ __launch_bounds__(128)
void gemm_wmma_kernel(const unsigned short* __restrict__ A,
                      const unsigned short* __restrict__ W,
                      const float* __restrict__ bias,
                      float* __restrict__ outF,
                      unsigned short* __restrict__ qo,
                      unsigned short* __restrict__ ko,
                      unsigned short* __restrict__ vo,
                      int M, int Nout, int K, int mode)
{
    // per buffer: A tile 128x32 bf16 (8KB) + B tile 128x32 bf16 (8KB)
    __shared__ unsigned short lds[2 * 8192];          // 32 KB, double buffered

    const int tid  = threadIdx.x;
    const int lane = tid & 31;
    const int wave = tid >> 5;
    const int m0   = blockIdx.x * 128;
    const int n0   = blockIdx.y * 128;
    const int wm   = (wave & 1) * 64;
    const int wn   = (wave >> 1) * 64;
    const int half = lane >> 4;
    const int ln   = lane & 15;

    const unsigned lds_base = (unsigned)(size_t)(&lds[0]);
    const int ksteps = K >> 5;

    f32x8 acc[4][4] = {};

    if (wave == 0) {                                  // prologue: stage step 0
        tdm_load_tile_2d(lds_base,        A + (size_t)m0 * K, 128, 32, M, K);
        tdm_load_tile_2d(lds_base + 8192, W + (size_t)n0 * K, 128, 32, Nout, K);
    }

    for (int s = 0; s < ksteps; ++s) {
        if (wave == 0) {
            if (s + 1 < ksteps) {                     // prefetch next K-slab
                unsigned boff = lds_base + ((s + 1) & 1) * 16384;
                tdm_load_tile_2d(boff,        A + (size_t)m0 * K + (s + 1) * 32,
                                 128, 32, M, K);
                tdm_load_tile_2d(boff + 8192, W + (size_t)n0 * K + (s + 1) * 32,
                                 128, 32, Nout, K);
                __builtin_amdgcn_s_wait_tensorcnt(2); // current pair landed
            } else {
                __builtin_amdgcn_s_wait_tensorcnt(0);
            }
        }
        __syncthreads();                              // publish buffer s&1

        const unsigned int* LA32 =
            (const unsigned int*)(lds + (s & 1) * 8192);
        const unsigned int* LB32 =
            (const unsigned int*)(lds + (s & 1) * 8192 + 4096);

        Frag af[4], bf[4];
        // A fragment: row = m, k = 16*half + i; one 32B contiguous run
        #pragma unroll
        for (int mt = 0; mt < 4; ++mt) {
            int row = wm + mt * 16 + ln;
            int kb  = 16 * half;
            #pragma unroll
            for (int j = 0; j < 8; ++j) {
                int e = row * 32 + kb + 2 * j;
                af[mt].u[j] = LA32[e >> 1];
            }
        }
        // B fragment: n = col, k = 16*half + i; one 32B contiguous run
        #pragma unroll
        for (int nt = 0; nt < 4; ++nt) {
            int row = wn + nt * 16 + ln;
            int kb  = 16 * half;
            #pragma unroll
            for (int j = 0; j < 8; ++j) {
                int e = row * 32 + kb + 2 * j;
                bf[nt].u[j] = LB32[e >> 1];
            }
        }
        #pragma unroll
        for (int mt = 0; mt < 4; ++mt)
            #pragma unroll
            for (int nt = 0; nt < 4; ++nt)
                acc[mt][nt] = __builtin_amdgcn_wmma_f32_16x16x32_bf16(
                    false, af[mt].v, false, bf[nt].v,
                    (short)0, acc[mt][nt], false, false);

        __syncthreads();                              // done reading buffer s&1
    }

    // epilogue
    for (int mt = 0; mt < 4; ++mt) {
        for (int nt = 0; nt < 4; ++nt) {
            int n = n0 + wn + nt * 16 + ln;
            float bv = bias[n];
            #pragma unroll
            for (int r = 0; r < 8; ++r) {
                int   m   = m0 + wm + mt * 16 + r + 8 * half;
                float val = acc[mt][nt][r] + bv;
                if (mode == 0) {
                    outF[(size_t)m * Nout + n] = val;
                } else {
                    // n in [0,3072): seg(q/k/v), head, hd; m = b*2048 + token
                    int seg = n >> 10, c = n & 1023;
                    int h = c >> 6, hd = c & 63;
                    int bb = m >> 11, nn = m & 2047;
                    unsigned short* dst = (seg == 0) ? qo : (seg == 1 ? ko : vo);
                    if (seg == 0) val *= 0.125f;      // fold softmax scale into Q
                    dst[(size_t)(((bb * 16 + h) * 2048) + nn) * 64 + hd] =
                        f32_to_bf16(val);
                }
            }
        }
    }
}

// ---------------------------------------------------------------------------
// Flash attention over bf16 Q/K/V [B,H,N=2048,hd=64] -> ctx bf16 [B,N,1024]
// grid: x = N/64 query blocks, y = B*H. 128 threads = 4 waves,
// each wave owns a 16-row query strip. Key blocks of 64, online softmax.
// ---------------------------------------------------------------------------
__global__ __launch_bounds__(128)
void flash_attn_kernel(const unsigned short* __restrict__ Q,
                       const unsigned short* __restrict__ K,
                       const unsigned short* __restrict__ V,
                       unsigned short* __restrict__ ctx)
{
    const int lane = threadIdx.x & 31;
    const int wave = threadIdx.x >> 5;
    const int half = lane >> 4;
    const int ln   = lane & 15;
    const int bh   = blockIdx.y;              // b*16 + h
    const int b    = bh >> 4;
    const int h    = bh & 15;
    const int q0   = blockIdx.x * 64 + wave * 16;

    __shared__ unsigned short ldsP[4][16 * 64];   // per-wave P strip (bf16)

    const unsigned short* Qh = Q + (size_t)bh * 2048 * 64;
    const unsigned short* Kh = K + (size_t)bh * 2048 * 64;
    const unsigned short* Vh = V + (size_t)bh * 2048 * 64;
    const unsigned int* Qh32 = (const unsigned int*)Qh;
    const unsigned int* Kh32 = (const unsigned int*)Kh;

    // Q A-fragments for the two 32-wide hd slices (Q pre-scaled by 0.125)
    Frag qf[2];
    {
        int row = q0 + ln;
        int kb  = 16 * half;
        #pragma unroll
        for (int kk = 0; kk < 2; ++kk)
            #pragma unroll
            for (int j = 0; j < 8; ++j) {
                int e = row * 64 + kk * 32 + kb + 2 * j;
                qf[kk].u[j] = Qh32[e >> 1];
            }
    }

    f32x8 O[4] = {};
    float rmax[8], rsum[8];
    #pragma unroll
    for (int r = 0; r < 8; ++r) { rmax[r] = -1e30f; rsum[r] = 0.0f; }

    for (int kb0 = 0; kb0 < 2048; kb0 += 64) {
        // ---- S = Q @ K^T for a 16x64 score strip (4 key tiles) ----
        f32x8 S[4] = {};
        #pragma unroll
        for (int kk = 0; kk < 2; ++kk) {
            #pragma unroll
            for (int t = 0; t < 4; ++t) {
                Frag kf;                       // B frag: n=key, k=hd (contig)
                int key = kb0 + t * 16 + ln;
                int cb  = kk * 32 + 16 * half;
                #pragma unroll
                for (int j = 0; j < 8; ++j)
                    kf.u[j] = Kh32[(key * 64 + cb + 2 * j) >> 1];
                S[t] = __builtin_amdgcn_wmma_f32_16x16x32_bf16(
                    false, qf[kk].v, false, kf.v, (short)0, S[t], false, false);
            }
        }

        // ---- online softmax (rows live per (r,half); 16-lane reductions) ----
        #pragma unroll
        for (int r = 0; r < 8; ++r) {
            float mx = fmaxf(fmaxf(S[0][r], S[1][r]), fmaxf(S[2][r], S[3][r]));
            #pragma unroll
            for (int d = 8; d >= 1; d >>= 1)
                mx = fmaxf(mx, __shfl_xor(mx, d, 32));
            float mnew  = fmaxf(rmax[r], mx);
            float alpha = __expf(rmax[r] - mnew);
            rmax[r] = mnew;
            #pragma unroll
            for (int t = 0; t < 4; ++t) O[t][r] *= alpha;
            float ps = 0.0f;
            #pragma unroll
            for (int t = 0; t < 4; ++t) {
                float p = __expf(S[t][r] - mnew);
                S[t][r] = p;
                ps += p;
            }
            #pragma unroll
            for (int d = 8; d >= 1; d >>= 1)
                ps += __shfl_xor(ps, d, 32);
            rsum[r] = rsum[r] * alpha + ps;
        }

        // ---- P -> LDS (bf16), re-read in A-fragment layout ----
        unsigned short* pw = ldsP[wave];
        #pragma unroll
        for (int t = 0; t < 4; ++t)
            #pragma unroll
            for (int r = 0; r < 8; ++r)
                pw[(r + 8 * half) * 64 + t * 16 + ln] = f32_to_bf16(S[t][r]);
        __syncthreads();

        const unsigned int* pw32 = (const unsigned int*)pw;
        #pragma unroll
        for (int kk = 0; kk < 2; ++kk) {
            Frag pf;                           // A frag of P: row=ln, k=key
            {
                int kb = kk * 32 + 16 * half;
                #pragma unroll
                for (int j = 0; j < 8; ++j) {
                    int e = ln * 64 + kb + 2 * j;
                    pf.u[j] = pw32[e >> 1];
                }
            }
            #pragma unroll
            for (int t = 0; t < 4; ++t) {
                Frag vf;                       // B frag of V: n=hd, k=key (strided)
                int hd = t * 16 + ln;
                int kb = kb0 + kk * 32 + 16 * half;
                #pragma unroll
                for (int i = 0; i < 16; ++i)
                    vf.s[i] = Vh[(size_t)(kb + i) * 64 + hd];
                O[t] = __builtin_amdgcn_wmma_f32_16x16x32_bf16(
                    false, pf.v, false, vf.v, (short)0, O[t], false, false);
            }
        }
        __syncthreads();
    }

    // ---- normalize and write ctx [B, N, H*64] in bf16 ----
    #pragma unroll
    for (int t = 0; t < 4; ++t)
        #pragma unroll
        for (int r = 0; r < 8; ++r) {
            int m  = q0 + r + 8 * half;
            int hd = t * 16 + ln;
            float o = O[t][r] / rsum[r];
            ctx[((size_t)b * 2048 + m) * 1024 + h * 64 + hd] = f32_to_bf16(o);
        }
}

// ---------------------------------------------------------------------------
extern "C" void kernel_launch(void* const* d_in, const int* in_sizes, int n_in,
                              void* d_out, int out_size, void* d_ws, size_t ws_size,
                              hipStream_t stream) {
    const float* x      = (const float*)d_in[0];
    const float* qkv_w  = (const float*)d_in[1];
    const float* qkv_b  = (const float*)d_in[2];
    const float* proj_w = (const float*)d_in[3];
    const float* proj_b = (const float*)d_in[4];
    float* out = (float*)d_out;

    const int B = 2, N = 2048, EMB = 1024, H = 16;
    const int M = B * N;                       // 4096

    // workspace carve-out (~50 MB total, bf16 buffers)
    size_t off = 0;
    auto carve = [&](size_t elems) {
        unsigned short* p = (unsigned short*)((char*)d_ws + off);
        off += ((elems * 2 + 255) / 256) * 256;
        return p;
    };
    unsigned short* xb    = carve((size_t)M * EMB);
    unsigned short* wqkvb = carve((size_t)3 * EMB * EMB);
    unsigned short* wprjb = carve((size_t)EMB * EMB);
    unsigned short* qb    = carve((size_t)B * H * N * 64);
    unsigned short* kb    = carve((size_t)B * H * N * 64);
    unsigned short* vb    = carve((size_t)B * H * N * 64);
    unsigned short* ctxb  = carve((size_t)M * EMB);

    // 1) f32 -> bf16 conversions
    {
        int n = M * EMB;
        cvt_bf16_kernel<<<(n + 255) / 256, 256, 0, stream>>>(x, xb, n);
        n = 3 * EMB * EMB;
        cvt_bf16_kernel<<<(n + 255) / 256, 256, 0, stream>>>(qkv_w, wqkvb, n);
        n = EMB * EMB;
        cvt_bf16_kernel<<<(n + 255) / 256, 256, 0, stream>>>(proj_w, wprjb, n);
    }

    // 2) QKV projection: [4096,1024] @ [3072,1024]^T + b -> q/k/v [B,H,N,64]
    gemm_wmma_kernel<<<dim3(M / 128, 3 * EMB / 128), 128, 0, stream>>>(
        xb, wqkvb, qkv_b, nullptr, qb, kb, vb, M, 3 * EMB, EMB, 1);

    // 3) flash attention -> ctx [B,N,1024] bf16
    flash_attn_kernel<<<dim3(N / 64, B * H), 128, 0, stream>>>(qb, kb, vb, ctxb);

    // 4) output projection: [4096,1024] @ [1024,1024]^T + b -> f32 out
    gemm_wmma_kernel<<<dim3(M / 128, EMB / 128), 128, 0, stream>>>(
        ctxb, wprjb, proj_b, out, nullptr, nullptr, nullptr, M, EMB, EMB, 0);
}
